// PGNN_69398081569451
// MI455X (gfx1250) — matrix operand order repaired and checked by reading, not compiled, not run on hardware
//
#include <hip/hip_runtime.h>
#include <hip/hip_fp16.h>
#include <math.h>

typedef __attribute__((ext_vector_type(16))) _Float16 v16h;
typedef __attribute__((ext_vector_type(8)))  float    v8f;
typedef __attribute__((ext_vector_type(4)))  uint     v4u;

namespace {

constexpr int N_NODES = 20000;
constexpr int DIM     = 64;
constexpr int LEN     = 50;
constexpr int NGRP    = N_NODES / 16;   // 1250 16-node groups
constexpr int RWAVES  = 10;             // waves per reduce workgroup
constexpr int RBLKS   = NGRP / RWAVES;  // 125 workgroups (exact)

// ---------------- workspace layout (bytes) ----------------
constexpr size_t OFF_UHID  = 0;                                        // f32 [N][64]
constexpr size_t OFF_IHID  = OFF_UHID  + (size_t)N_NODES*DIM*4;
constexpr size_t OFF_UHIDH = OFF_IHID  + (size_t)N_NODES*DIM*4;        // f16 [N][64]
constexpr size_t OFF_IHIDH = OFF_UHIDH + (size_t)N_NODES*DIM*2;
constexpr size_t OFF_UAGG  = OFF_IHIDH + (size_t)N_NODES*DIM*2;        // f32 [N][64]
constexpr size_t OFF_IAGG  = OFF_UAGG  + (size_t)N_NODES*DIM*4;
constexpr size_t OFF_GRUIN = OFF_IAGG  + (size_t)N_NODES*DIM*4;        // f16 [N][50][64]
constexpr size_t OFF_ATT   = OFF_GRUIN + (size_t)N_NODES*LEN*DIM*2;    // i32 [N][50]
constexpr size_t OFF_WB    = OFF_ATT   + (size_t)N_NODES*LEN*4;        // weight blob

// weight blob internal offsets (all B-operands stored WMMA-fragment-swizzled)
constexpr size_t WB_WUT  = 0;                     // f16 swizzled [64x64]  W_user^T
constexpr size_t WB_WIT  = WB_WUT + 64*64*2;      // f16 swizzled [64x64]  W_item^T
constexpr size_t WB_SIDE = WB_WIT + 64*64*2;      // two per-side blocks follow
constexpr size_t SB_GWIT = 0;                     // f16 swizzled [2][64x192] Wi^T
constexpr size_t SB_GWHT = SB_GWIT + 2*64*192*2;  // f16 swizzled [2][64x192] Wh^T
constexpr size_t SB_GLIN = SB_GWHT + 2*64*192*2;  // f16 swizzled [128x64]
constexpr size_t SB_AGGT = SB_GLIN + 128*64*2;    // f16 swizzled [128x64]
constexpr size_t SB_UPDT = SB_AGGT + 128*64*2;    // f16 swizzled [128x64]
constexpr size_t SB_BSUM = SB_UPDT + 128*64*2;    // f32 [2][128]  bi+bh (r,z)
constexpr size_t SB_BIN  = SB_BSUM + 2*128*4;     // f32 [2][64]   bi (n)
constexpr size_t SB_BHN  = SB_BIN  + 2*64*4;      // f32 [2][64]   bh (n)
constexpr size_t SB_TEH  = SB_BHN  + 2*64*4;      // f16 [50][64]  te
constexpr size_t SB_TEK  = SB_TEH  + 50*64*2;     // f16 [50][64]  te_k
constexpr size_t SB_SIZE = SB_TEK  + 50*64*2;
constexpr size_t OFF_FSEQ = OFF_WB + WB_SIDE + 2*SB_SIZE;  // f16 [1250][50][16][64] fwd hidden

// reduce-kernel LDS: shared weights once per workgroup + small per-wave scratch
constexpr size_t SH_BYTES =
    (size_t)(2*(2*DIM*192) + 2*(128*DIM) + 2*(LEN*DIM)) * 2   // Wi,Wh,glin,agg,te,tek (f16)
    + 512 * 4;                                                // biases (f32)
constexpr size_t PW_BYTES =
    (size_t)(3 * 16 * DIM) * 2       // sH, sX, sLa (f16)  [sH+sX overlay as cat buffer]
    + (size_t)(16 * DIM) * 4         // sDst (f32)
    + (size_t)(16 * LEN) * 4;        // sAtt (i32)
constexpr size_t RED_SMEM = SH_BYTES + (size_t)RWAVES * PW_BYTES;  // ~280 KB < 320 KB

// ---------------- utilities ----------------
__device__ __forceinline__ void wave_sync() {
#if defined(__has_builtin)
#if __has_builtin(__builtin_amdgcn_wave_barrier)
  __builtin_amdgcn_wave_barrier();   // scheduling fence; LDS is in-order within a wave
#else
  __syncthreads();
#endif
#else
  __syncthreads();
#endif
}

__device__ __forceinline__ void copy16n(void* dst, const void* src, int nchunks,
                                        int tid, int stride) {
  v4u* d = (v4u*)dst;
  const v4u* s = (const v4u*)src;
  for (int i = tid; i < nchunks; i += stride) d[i] = s[i];
}

// ---------------- WMMA fragment helpers (wave32, 16x16x32 f16) ----------------
// A layout (16x32, 16-bit): lanes 0-15 -> M=lane, K in {kb..kb+7, kb+16..kb+23}, kb=(lane>>4)*8
// Both halves are 16B-contiguous -> 2x b128 loads per lane (LDS or global).
__device__ __forceinline__ v16h load_a_frag(const _Float16* base, int ld, int k0) {
  int lane = threadIdx.x & 31;
  int row  = lane & 15;
  int kb   = (lane >> 4) * 8;
  const uint* p0 = (const uint*)(base + row * ld + k0 + kb);
  const uint* p1 = (const uint*)(base + row * ld + k0 + kb + 16);
  v16h a;
  uint* ap = (uint*)&a;
#pragma unroll
  for (int i = 0; i < 4; ++i) ap[i] = p0[i];
#pragma unroll
  for (int i = 0; i < 4; ++i) ap[4 + i] = p1[i];
  return a;
}

// B operand pre-swizzled fragment-major: frag (kt,nt) of a [K x N] operand stores the
// 16 halfs lane l needs contiguously at (kt*nb + nt)*512 + l*16  -> 2x ds_load_b128.
__device__ __forceinline__ v16h load_b_sw(const _Float16* base, int kt, int nt, int nb) {
  int lane = threadIdx.x & 31;
  const uint* p = (const uint*)(base + (size_t)((kt * nb + nt) * 512 + lane * 16));
  v16h b;
  uint* bp = (uint*)&b;
#pragma unroll
  for (int i = 0; i < 8; ++i) bp[i] = p[i];
  return b;
}

__device__ __forceinline__ v8f wmma16(v16h a, v16h b, v8f c) {
  return __builtin_amdgcn_wmma_f32_16x16x32_f16(false, a, false, b, (short)0, c, false, false);
}

__device__ __forceinline__ float sigmoidf_(float x) { return 1.f / (1.f + __expf(-x)); }

// ---------------- kernel: weight prep (transpose + f16 + B-fragment swizzle) ----------------
// Produces swizzled B for:  out = in @ W^T  with W row-major [N][K]:  B[k][n] = W[n][k]
__device__ void swizzleB(_Float16* dst, const float* w, int K, int N, int tid, int nthreads) {
  int nb = N / 16;
  int total = K * N;
  for (int e = tid; e < total; e += nthreads) {
    int f = e >> 9;                 // fragment index (512 halfs each)
    int r = e & 511;
    int lane = r >> 4, i = r & 15;
    int kt = f / nb, nt = f % nb;
    int k = kt * 32 + (lane >> 4) * 8 + (i & 7) + (i >> 3) * 16;
    int n = nt * 16 + (lane & 15);
    dst[e] = (_Float16)w[(size_t)n * K + k];
  }
}

struct PrepParams {
  const float *Wu, *Wi_;
  const float *u_te, *u_tek, *i_te, *i_tek;
  const float *guWi, *guWh, *gubi, *gubh;
  const float *giWi, *giWh, *gibi, *gibh;
  const float *glu, *gli, *aggu, *aggi, *updu, *updi;
  char* ws;
};

__global__ void k_prep(PrepParams P) {
  int tid = threadIdx.x;  // 256
  swizzleB((_Float16*)(P.ws + OFF_WB + WB_WUT), P.Wu, 64, 64, tid, 256);
  swizzleB((_Float16*)(P.ws + OFF_WB + WB_WIT), P.Wi_, 64, 64, tid, 256);
  for (int s = 0; s < 2; ++s) {  // s==0: item side ('i' weights), s==1: user side ('u')
    char* base = P.ws + OFF_WB + WB_SIDE + (size_t)s * SB_SIZE;
    const float* gWi = s ? P.guWi : P.giWi;
    const float* gWh = s ? P.guWh : P.giWh;
    const float* gbi = s ? P.gubi : P.gibi;
    const float* gbh = s ? P.gubh : P.gibh;
    const float* gl  = s ? P.glu  : P.gli;
    const float* ag  = s ? P.aggu : P.aggi;
    const float* up  = s ? P.updu : P.updi;
    const float* te  = s ? P.u_te  : P.i_te;
    const float* tk  = s ? P.u_tek : P.i_tek;
    for (int d = 0; d < 2; ++d) {
      swizzleB((_Float16*)(base + SB_GWIT) + d * 64 * 192, gWi + (size_t)d * 192 * 64, 64, 192, tid, 256);
      swizzleB((_Float16*)(base + SB_GWHT) + d * 64 * 192, gWh + (size_t)d * 192 * 64, 64, 192, tid, 256);
    }
    swizzleB((_Float16*)(base + SB_GLIN), gl, 128, 64, tid, 256);
    swizzleB((_Float16*)(base + SB_AGGT), ag, 128, 64, tid, 256);
    swizzleB((_Float16*)(base + SB_UPDT), up, 128, 64, tid, 256);
    float* bsum = (float*)(base + SB_BSUM);
    for (int i = tid; i < 256; i += 256) {
      int d = i >> 7, c = i & 127;
      bsum[i] = gbi[d * 192 + c] + gbh[d * 192 + c];
    }
    float* bin = (float*)(base + SB_BIN);
    float* bhn = (float*)(base + SB_BHN);
    for (int i = tid; i < 128; i += 256) {
      int d = i >> 6, c = i & 63;
      bin[i] = gbi[d * 192 + 128 + c];
      bhn[i] = gbh[d * 192 + 128 + c];
    }
    _Float16* teh = (_Float16*)(base + SB_TEH);
    _Float16* tkh = (_Float16*)(base + SB_TEK);
    for (int i = tid; i < 50 * 64; i += 256) {
      teh[i] = (_Float16)te[i];
      tkh[i] = (_Float16)tk[i];
    }
  }
}

// ---------------- kernel: hid = feat @ W^T (WMMA), f32 + f16 outputs ----------------
__global__ void k_proj(const float* __restrict__ feat, const _Float16* __restrict__ WTsw,
                       float* __restrict__ hid_f32, _Float16* __restrict__ hid_h) {
  __shared__ _Float16 sA[16 * 64];
  __shared__ _Float16 sW[64 * 64];
  int lane = threadIdx.x;
  int row0 = blockIdx.x * 16;
  for (int i = lane; i < 16 * 64; i += 32)
    sA[i] = (_Float16)feat[(size_t)(row0 + (i >> 6)) * 64 + (i & 63)];
  copy16n(sW, WTsw, 64 * 64 / 8, lane, 32);
  __syncthreads();
  v16h a0 = load_a_frag(sA, 64, 0);
  v16h a1 = load_a_frag(sA, 64, 32);
  int col = lane & 15, rb = (lane >> 4) * 8;
#pragma unroll
  for (int t = 0; t < 4; ++t) {
    v8f c = {};
    c = wmma16(a0, load_b_sw(sW, 0, t, 4), c);
    c = wmma16(a1, load_b_sw(sW, 1, t, 4), c);
#pragma unroll
    for (int r = 0; r < 8; ++r) {
      size_t o = (size_t)(row0 + rb + r) * 64 + t * 16 + col;
      hid_f32[o] = c[r];
      hid_h[o]   = (_Float16)c[r];
    }
  }
}

// ---------------- kernel: per-node time sort + mailbox gather (f16, b128) ----------------
__global__ void k_sortgather(const int* __restrict__ nbr_idx, const int* __restrict__ nbr_time,
                             const _Float16* __restrict__ src_h,
                             _Float16* __restrict__ gru_in, int* __restrict__ attidx) {
  int n = blockIdx.x;
  __shared__ int inv[LEN];   // inv[rank] = original position
  __shared__ int sidx[LEN];
  int tid = threadIdx.x;     // blockDim = 64
  const int* tp = nbr_time + (size_t)n * LEN;
  const int* ip = nbr_idx  + (size_t)n * LEN;
  if (tid < LEN) {
    int t_me = tp[tid];
    int rank = 0;
    for (int j = 0; j < LEN; ++j) {
      int tj = tp[j];
      rank += (tj < t_me) || (tj == t_me && j < tid);   // stable rank
    }
    inv[rank] = tid;
    attidx[(size_t)n * LEN + tid] = LEN - 1 - rank;     // reversed rank at original pos
    sidx[tid] = ip[tid];
  }
  __syncthreads();
  const v4u* srcw = (const v4u*)src_h;                   // 8 x b128 per 64-f16 row
  v4u* dstw = (v4u*)(gru_in + (size_t)n * LEN * DIM);
  for (int q = tid; q < LEN * 8; q += 64) {
    int s = q >> 3, c = q & 7;
    int row = sidx[inv[s]];
    dstw[s * 8 + c] = srcw[(size_t)row * 8 + c];
  }
}

// ---------------- fused biGRU scan + glin + attention + agg ----------------
struct ReduceParams {
  const _Float16* gru_in;  // [N][50][64] time-sorted mail (f16)
  const int*      attidx;  // [N][50]
  const float*    dst_hid; // [N][64]
  const char*     wb;      // per-side weight block (swizzled, see SB_*)
  _Float16*       fseq;    // [1250][50][16][64] forward hidden sequence (f16)
  float*          out;     // [N][64]
};

__device__ __forceinline__ void gru_step(int dir, const _Float16* sX, _Float16* sH,
                                         float* hC, const _Float16* sWi,
                                         const _Float16* sWh, const float* sB) {
  int lane = threadIdx.x & 31;
  v16h ax0 = load_a_frag(sX, 64, 0);
  v16h ax1 = load_a_frag(sX, 64, 32);
  v16h ah0 = load_a_frag(sH, 64, 0);
  v16h ah1 = load_a_frag(sH, 64, 32);
  const _Float16* Wi = sWi + dir * 64 * 192;   // 24 swizzled frags, nb = 12
  const _Float16* Wh = sWh + dir * 64 * 192;
  v8f rz[8], xn[4], gn[4];
#pragma unroll
  for (int j = 0; j < 8; ++j) {                 // r,z gates: xw+gh share one accumulator
    v8f c = {};
    c = wmma16(ax0, load_b_sw(Wi, 0, j, 12), c);
    c = wmma16(ax1, load_b_sw(Wi, 1, j, 12), c);
    c = wmma16(ah0, load_b_sw(Wh, 0, j, 12), c);
    c = wmma16(ah1, load_b_sw(Wh, 1, j, 12), c);
    rz[j] = c;
  }
#pragma unroll
  for (int j = 0; j < 4; ++j) {                 // n gate: xw and gh kept separate
    v8f c = {};
    c = wmma16(ax0, load_b_sw(Wi, 0, 8 + j, 12), c);
    c = wmma16(ax1, load_b_sw(Wi, 1, 8 + j, 12), c);
    xn[j] = c;
    v8f d = {};
    d = wmma16(ah0, load_b_sw(Wh, 0, 8 + j, 12), d);
    d = wmma16(ah1, load_b_sw(Wh, 1, 8 + j, 12), d);
    gn[j] = d;
  }
  const float* bs = sB + dir * 128;
  const float* bi = sB + 256 + dir * 64;
  const float* bh = sB + 384 + dir * 64;
  int col = lane & 15, rb = (lane >> 4) * 8;
  wave_sync();
#pragma unroll
  for (int ti = 0; ti < 4; ++ti) {
    int c0 = ti * 16 + col;
#pragma unroll
    for (int r = 0; r < 8; ++r) {
      float rv = sigmoidf_(rz[ti][r] + bs[c0]);
      float zv = sigmoidf_(rz[4 + ti][r] + bs[64 + c0]);
      float nv = tanhf(xn[ti][r] + bi[c0] + rv * (gn[ti][r] + bh[c0]));
      float hv = (1.f - zv) * nv + zv * hC[ti * 8 + r];
      hC[ti * 8 + r] = hv;
      sH[(rb + r) * 64 + c0] = (_Float16)hv;
    }
  }
  wave_sync();
}

__global__ void k_reduce(ReduceParams P) {
  extern __shared__ char smem_raw[];
  // --- block-shared region: weights/constants, loaded once per workgroup ---
  _Float16* sWi = (_Float16*)smem_raw;            // [2][64x192] Wi^T (then Wh^T)
  _Float16* sWh = sWi + 2 * DIM * 192;
  _Float16* sGl = sWh + 2 * DIM * 192;            // [128x64] glin^T
  _Float16* sAg = sGl + 128 * DIM;                // [128x64] agg^T
  _Float16* sTe = sAg + 128 * DIM;                // [50][64]
  _Float16* sTk = sTe + LEN * DIM;
  float*    sB  = (float*)(sTk + LEN * DIM);      // 512 floats of biases
  char*     pw  = (char*)(sB + 512);
  // --- per-wave private region ---
  int tid = threadIdx.x, lane = tid & 31, wid = tid >> 5;
  char* wbuf = pw + (size_t)wid * PW_BYTES;
  _Float16* sH  = (_Float16*)wbuf;                // [16][64] hidden state
  _Float16* sX  = sH + 16 * DIM;                  // [16][64] X_t staging (sH+sX = cat buf)
  _Float16* sLa = sX + 16 * DIM;                  // [16][64] last message
  float*    sDst = (float*)(sLa + 16 * DIM);      // [16][64]
  int*      sAtt = (int*)(sDst + 16 * DIM);       // [16][50]

  int group = blockIdx.x * RWAVES + wid;
  int node0 = group * 16;
  _Float16* Fseq = P.fseq + (size_t)group * LEN * 16 * DIM;

  // cooperative shared loads (all threads), blob regions are contiguous as LDS expects
  copy16n(sWi, P.wb + SB_GWIT, (2 * 2 * DIM * 192) / 8, tid, RWAVES * 32);
  copy16n(sGl, P.wb + SB_GLIN, (2 * 128 * DIM) / 8, tid, RWAVES * 32);
  copy16n(sTe, P.wb + SB_TEH, (2 * LEN * DIM) / 8, tid, RWAVES * 32);
  copy16n(sB, P.wb + SB_BSUM, (512 * 4) / 16, tid, RWAVES * 32);
  // per-wave loads
  copy16n(sAtt, P.attidx + (size_t)node0 * LEN, (16 * LEN) / 4, lane, 32);
  copy16n(sDst, P.dst_hid + (size_t)node0 * DIM, (16 * DIM) / 4, lane, 32);
  for (int i = lane; i < 16 * 8; i += 32) {       // last (most recent) message rows
    int m = i >> 3, c = i & 7;
    ((v4u*)sLa)[m * 8 + c] =
        ((const v4u*)(P.gru_in + ((size_t)(node0 + m) * LEN + (LEN - 1)) * DIM))[c];
  }
  for (int i = lane; i < 16 * DIM / 8; i += 32) ((v4u*)sH)[i] = (v4u){0, 0, 0, 0};
  __syncthreads();   // one real barrier: shared weights visible to all waves

  float hC[32];
#pragma unroll
  for (int i = 0; i < 32; ++i) hC[i] = 0.f;

  // ---- forward scan: t = 0..49, hidden tiles spilled to global fseq ----
  for (int t = 0; t < LEN; ++t) {
    for (int i = lane; i < 16 * 8; i += 32) {
      int m = i >> 3, c = i & 7;
      ((v4u*)sX)[m * 8 + c] =
          ((const v4u*)(P.gru_in + ((size_t)(node0 + m) * LEN + t) * DIM))[c];
    }
    if (t + 1 < LEN)
      __builtin_prefetch(&P.gru_in[((size_t)node0 * LEN + t + 1) * DIM], 0, 1);
    wave_sync();
    gru_step(0, sX, sH, hC, sWi, sWh, sB);
    for (int i = lane; i < 16 * 8; i += 32)      // spill f_t tile (2 KB) to global
      ((v4u*)(Fseq + t * (16 * DIM)))[i] = ((const v4u*)sH)[i];
  }

  // ---- backward scan fused with glin + online-softmax attention ----
  for (int i = lane; i < 16 * DIM / 8; i += 32) ((v4u*)sH)[i] = (v4u){0, 0, 0, 0};
#pragma unroll
  for (int i = 0; i < 32; ++i) hC[i] = 0.f;
  wave_sync();

  int col = lane & 15, rb = (lane >> 4) * 8;
  float mrun[8], srun[8], accA[32];
#pragma unroll
  for (int r = 0; r < 8; ++r) { mrun[r] = -1e30f; srun[r] = 0.f; }
#pragma unroll
  for (int i = 0; i < 32; ++i) accA[i] = 0.f;

  for (int t = LEN - 1; t >= 0; --t) {
    for (int i = lane; i < 16 * 8; i += 32) {
      int m = i >> 3, c = i & 7;
      ((v4u*)sX)[m * 8 + c] =
          ((const v4u*)(P.gru_in + ((size_t)(node0 + m) * LEN + t) * DIM))[c];
    }
    if (t > 0) {
      __builtin_prefetch(&P.gru_in[((size_t)node0 * LEN + t - 1) * DIM], 0, 1);
      __builtin_prefetch(Fseq + (t - 1) * (16 * DIM), 0, 1);
    }
    wave_sync();
    gru_step(1, sX, sH, hC, sWi, sWh, sB);   // sH now holds b_t

    const _Float16* Ft = Fseq + t * (16 * DIM);  // f_t from global (b128 A-frag loads)
    v16h af0 = load_a_frag(Ft, 64, 0);
    v16h af1 = load_a_frag(Ft, 64, 32);
    v16h ab0 = load_a_frag(sH, 64, 0);
    v16h ab1 = load_a_frag(sH, 64, 32);
    v8f g[4];
#pragma unroll
    for (int j = 0; j < 4; ++j) {            // g_t = [f_t, b_t] @ glin^T  (K=128, nb=4)
      v8f c = {};
      c = wmma16(af0, load_b_sw(sGl, 0, j, 4), c);
      c = wmma16(af1, load_b_sw(sGl, 1, j, 4), c);
      c = wmma16(ab0, load_b_sw(sGl, 2, j, 4), c);
      c = wmma16(ab1, load_b_sw(sGl, 3, j, 4), c);
      g[j] = c;
    }
    // e[m] = dot(te[attidx]+g, dst_hid)/8  (cross-lane reduce over 16 cols per half-wave)
    float part[8];
#pragma unroll
    for (int r = 0; r < 8; ++r) {
      int m = rb + r;
      int a = sAtt[m * LEN + t];
      float p = 0.f;
#pragma unroll
      for (int ti = 0; ti < 4; ++ti) {
        int c0 = ti * 16 + col;
        p += ((float)sTe[a * 64 + c0] + g[ti][r]) * sDst[m * 64 + c0];
      }
      part[r] = p;
    }
#pragma unroll
    for (int off = 1; off < 16; off <<= 1) {
#pragma unroll
      for (int r = 0; r < 8; ++r) part[r] += __shfl_xor(part[r], off, 32);
    }
    // online softmax accumulation of alpha*(g + te_k[attidx])
#pragma unroll
    for (int r = 0; r < 8; ++r) {
      int m = rb + r;
      float e  = part[r] * 0.125f;
      float mn = fmaxf(mrun[r], e);
      float sc = __expf(mrun[r] - mn);
      float w  = __expf(e - mn);
      srun[r] = srun[r] * sc + w;
      mrun[r] = mn;
      int a = sAtt[m * LEN + t];
#pragma unroll
      for (int ti = 0; ti < 4; ++ti) {
        int c0 = ti * 16 + col;
        accA[ti * 8 + r] = accA[ti * 8 + r] * sc + w * (g[ti][r] + (float)sTk[a * 64 + c0]);
      }
    }
  }

  // ---- finalize: out = [hid, last_em] @ agg^T  (cat buffer overlays sH+sX) ----
  wave_sync();
  _Float16* sCat = sH;  // [16][128]
#pragma unroll
  for (int ti = 0; ti < 4; ++ti) {
#pragma unroll
    for (int r = 0; r < 8; ++r) {
      int m = rb + r, c0 = ti * 16 + col;
      sCat[m * 128 + c0] = (_Float16)(accA[ti * 8 + r] / srun[r]);
    }
  }
  for (int i = lane; i < 16 * 8; i += 32) {
    int m = i >> 3, c = i & 7;
    ((v4u*)(sCat + m * 128 + 64))[c] = ((const v4u*)(sLa + m * 64))[c];
  }
  wave_sync();
  v16h ac0 = load_a_frag(sCat, 128, 0);
  v16h ac1 = load_a_frag(sCat, 128, 32);
  v16h ac2 = load_a_frag(sCat, 128, 64);
  v16h ac3 = load_a_frag(sCat, 128, 96);
#pragma unroll
  for (int j = 0; j < 4; ++j) {
    v8f c = {};
    c = wmma16(ac0, load_b_sw(sAg, 0, j, 4), c);
    c = wmma16(ac1, load_b_sw(sAg, 1, j, 4), c);
    c = wmma16(ac2, load_b_sw(sAg, 2, j, 4), c);
    c = wmma16(ac3, load_b_sw(sAg, 3, j, 4), c);
#pragma unroll
    for (int r = 0; r < 8; ++r)
      P.out[(size_t)(node0 + rb + r) * DIM + j * 16 + col] = c[r];
  }
}

// ---------------- kernel: new = tanh([agg, feat] @ upd^T) ----------------
__global__ void k_update(const float* __restrict__ agg, const float* __restrict__ feat,
                         const _Float16* __restrict__ updTsw, float* __restrict__ out) {
  __shared__ _Float16 sA[16 * 128];
  __shared__ _Float16 sW[128 * 64];
  int lane = threadIdx.x;
  int row0 = blockIdx.x * 16;
  for (int i = lane; i < 16 * 64; i += 32) {
    int m = i >> 6, c = i & 63;
    sA[m * 128 + c]      = (_Float16)agg[(size_t)(row0 + m) * 64 + c];
    sA[m * 128 + 64 + c] = (_Float16)feat[(size_t)(row0 + m) * 64 + c];
  }
  copy16n(sW, updTsw, 128 * 64 / 8, lane, 32);
  __syncthreads();
  v16h a0 = load_a_frag(sA, 128, 0);
  v16h a1 = load_a_frag(sA, 128, 32);
  v16h a2 = load_a_frag(sA, 128, 64);
  v16h a3 = load_a_frag(sA, 128, 96);
  int col = lane & 15, rb = (lane >> 4) * 8;
#pragma unroll
  for (int j = 0; j < 4; ++j) {
    v8f c = {};
    c = wmma16(a0, load_b_sw(sW, 0, j, 4), c);
    c = wmma16(a1, load_b_sw(sW, 1, j, 4), c);
    c = wmma16(a2, load_b_sw(sW, 2, j, 4), c);
    c = wmma16(a3, load_b_sw(sW, 3, j, 4), c);
#pragma unroll
    for (int r = 0; r < 8; ++r)
      out[(size_t)(row0 + rb + r) * 64 + j * 16 + col] = tanhf(c[r]);
  }
}

}  // namespace

extern "C" void kernel_launch(void* const* d_in, const int* in_sizes, int n_in,
                              void* d_out, int out_size, void* d_ws, size_t ws_size,
                              hipStream_t stream) {
  (void)in_sizes; (void)n_in; (void)out_size; (void)ws_size;
  const float* user_feat = (const float*)d_in[0];
  const float* item_feat = (const float*)d_in[1];
  const int* item_nbr_idx  = (const int*)d_in[2];
  const int* item_nbr_time = (const int*)d_in[3];
  const int* user_nbr_idx  = (const int*)d_in[4];
  const int* user_nbr_time = (const int*)d_in[5];

  char* ws = (char*)d_ws;
  float*     uhid  = (float*)(ws + OFF_UHID);
  float*     ihid  = (float*)(ws + OFF_IHID);
  _Float16*  uhidh = (_Float16*)(ws + OFF_UHIDH);
  _Float16*  ihidh = (_Float16*)(ws + OFF_IHIDH);
  float*     uagg  = (float*)(ws + OFF_UAGG);
  float*     iagg  = (float*)(ws + OFF_IAGG);
  _Float16*  gruin = (_Float16*)(ws + OFF_GRUIN);
  int*       att   = (int*)(ws + OFF_ATT);
  _Float16*  fseq  = (_Float16*)(ws + OFF_FSEQ);
  char*      blob  = ws + OFF_WB;

  // 1) weight prep (transpose, f16 convert, WMMA B-fragment swizzle)
  PrepParams pp;
  pp.Wu = (const float*)d_in[6];  pp.Wi_ = (const float*)d_in[7];
  pp.u_te = (const float*)d_in[8];  pp.u_tek = (const float*)d_in[9];
  pp.i_te = (const float*)d_in[10]; pp.i_tek = (const float*)d_in[11];
  pp.guWi = (const float*)d_in[12]; pp.guWh = (const float*)d_in[13];
  pp.gubi = (const float*)d_in[14]; pp.gubh = (const float*)d_in[15];
  pp.giWi = (const float*)d_in[16]; pp.giWh = (const float*)d_in[17];
  pp.gibi = (const float*)d_in[18]; pp.gibh = (const float*)d_in[19];
  pp.glu  = (const float*)d_in[20]; pp.gli  = (const float*)d_in[21];
  pp.aggu = (const float*)d_in[22]; pp.aggi = (const float*)d_in[23];
  pp.updu = (const float*)d_in[24]; pp.updi = (const float*)d_in[25];
  pp.ws = ws;
  k_prep<<<1, 256, 0, stream>>>(pp);

  // 2) input projections (WMMA)
  k_proj<<<NGRP, 32, 0, stream>>>(user_feat, (const _Float16*)(blob + WB_WUT), uhid, uhidh);
  k_proj<<<NGRP, 32, 0, stream>>>(item_feat, (const _Float16*)(blob + WB_WIT), ihid, ihidh);

  // 3) item reduce ('by' edges: user -> item), side 0 weights
  k_sortgather<<<N_NODES, 64, 0, stream>>>(item_nbr_idx, item_nbr_time, uhidh, gruin, att);
  {
    ReduceParams rp;
    rp.gru_in = gruin; rp.attidx = att; rp.dst_hid = ihid;
    rp.wb = blob + WB_SIDE + (size_t)0 * SB_SIZE;
    rp.fseq = fseq; rp.out = iagg;
    k_reduce<<<RBLKS, RWAVES * 32, RED_SMEM, stream>>>(rp);
  }

  // 4) user reduce ('pby' edges: item -> user), side 1 weights
  k_sortgather<<<N_NODES, 64, 0, stream>>>(user_nbr_idx, user_nbr_time, ihidh, gruin, att);
  {
    ReduceParams rp;
    rp.gru_in = gruin; rp.attidx = att; rp.dst_hid = uhid;
    rp.wb = blob + WB_SIDE + (size_t)1 * SB_SIZE;
    rp.fseq = fseq; rp.out = uagg;
    k_reduce<<<RBLKS, RWAVES * 32, RED_SMEM, stream>>>(rp);
  }

  // 5) final updates: user_new then item_new (concatenated in d_out)
  float* out_f = (float*)d_out;
  k_update<<<NGRP, 32, 0, stream>>>(uagg, user_feat,
      (const _Float16*)(blob + WB_SIDE + (size_t)1 * SB_SIZE + SB_UPDT), out_f);
  k_update<<<NGRP, 32, 0, stream>>>(iagg, item_feat,
      (const _Float16*)(blob + WB_SIDE + (size_t)0 * SB_SIZE + SB_UPDT),
      out_f + (size_t)N_NODES * DIM);
}
